// LinearMultiHeadAttentionNonCausal_47064251629940
// MI455X (gfx1250) — matrix operand order repaired and checked by reading, not compiled, not run on hardware
//
#include <hip/hip_runtime.h>
#include <hip/hip_bf16.h>

// Linear multi-head attention (non-causal), fp32 in/out, f16 WMMA compute.
// B=4, S=8192, D_MODEL=512, H=8, DEPTH=64.
// Workspace (~111 MB):
//   WT      : 3*512*512 f16        weights transposed [n][k]
//   Qh      : [B][H][S][D] f16     phi(q@wq+bq)
//   KhT     : [B][H][D][S] f16     phi(k@wk+bk), transposed
//   VhT     : [B][H][D][S] f16     v@wv+bv, transposed
//   kvT     : [B*H][E][D] f16      kv state, transposed
//   kred    : [B*H][D] f16         sum_s kh + EPS
//   kvPart  : [16][B*H][E][D] f32  per-S-chunk partial kv
//   kredPart: [16][B*H][D] f32     per-S-chunk partial k_red

typedef __attribute__((ext_vector_type(16))) _Float16 v16h;
typedef __attribute__((ext_vector_type(8)))  _Float16 v8h;
typedef __attribute__((ext_vector_type(8)))  float    v8f;
typedef __attribute__((ext_vector_type(4)))  float    v4f;

#define D_MODEL 512
#define NUM_HEADS 8
#define DEPTH 64
#define SEQ 8192
#define BATCH 4
#define MTOT (BATCH*SEQ)      // 32768
#define NCHUNK 16             // S split for the kv reduction
#define SCHUNK (SEQ/NCHUNK)   // 512
#define EPS_F 1e-8f

// phi(x) = elu(x)+1 = x>0 ? x+1 : exp(x)   (branchless: exp + cndmask)
static __device__ __forceinline__ float phi_f(float v) {
  const float e = __expf(fminf(v, 0.0f));
  return (v > 0.0f) ? (v + 1.0f) : e;
}

// Load a 16-bit A-fragment (16x32, row-major, K contiguous) for this lane.
// halves [0..15] = K(base..base+15), base = lhalf*16.
static __device__ __forceinline__ v16h wmma_ld_a16(const _Float16* rowk, int lhalf) {
  const v8h lo = *reinterpret_cast<const v8h*>(rowk + lhalf * 16);
  const v8h hi = *reinterpret_cast<const v8h*>(rowk + lhalf * 16 + 8);
  return __builtin_shufflevector(lo, hi, 0,1,2,3,4,5,6,7,8,9,10,11,12,13,14,15);
}

// ---------------------------------------------------------------------------
// Kernel 0: transpose + convert weights to f16:  WT[z][n][k] = W_z[k][n]
// ---------------------------------------------------------------------------
__global__ void lmha_wprep(const float* __restrict__ wq, const float* __restrict__ wk,
                           const float* __restrict__ wv, _Float16* __restrict__ WT) {
  const int idx = blockIdx.x * blockDim.x + threadIdx.x;   // 0 .. 3*512*512-1
  const int z   = idx >> 18;
  const int rem = idx & 262143;
  const int k   = rem & 511;
  const int n   = rem >> 9;
  const float* W = (z == 0) ? wq : (z == 1) ? wk : wv;
  WT[idx] = (_Float16)W[(size_t)k * D_MODEL + n];
}

// ---------------------------------------------------------------------------
// Kernel 1: LDS-tiled projection GEMM + bias + (optional) phi + f16 store.
// Block = 256 threads (8 waves), macro-tile 128m x 128n, K-step 32.
// A (fp32->f16) and B (f16) staged in LDS each K-step; wave w owns m-tile w
// and all 8 n-tiles -> one A fragment feeds 8 WMMAs per K-step.
//   MODE 0: Q -> dst[b][h][s][d] (phi)
//   MODE 1: K -> dst[b][h][d][s] (phi, transposed v8h store)
//   MODE 2: V -> dst[b][h][d][s] (no phi, transposed v8h store)
// ---------------------------------------------------------------------------
template <int MODE>
__global__ void lmha_proj(const float* __restrict__ X, const _Float16* __restrict__ WT,
                          const float* __restrict__ bias, _Float16* __restrict__ dst) {
  __shared__ __attribute__((aligned(32))) _Float16 lA[128 * 32];
  __shared__ __attribute__((aligned(32))) _Float16 lB[128 * 32];

  const int tid   = threadIdx.x;
  const int wave  = tid >> 5;
  const int lane  = tid & 31;
  const int lmod  = lane & 15;
  const int lhalf = lane >> 4;
  const int mbase = blockIdx.x * 128;        // 256 m-blocks
  const int nbase = blockIdx.y * 128;        // 4 n-blocks

  // staging roles
  const int srow = tid >> 1;                 // 0..127
  const int sseg = (tid & 1) * 16;           // k sub-offset 0 / 16

  const float*    xsrc = X  + (size_t)(mbase + srow) * D_MODEL + sseg;
  const _Float16* wsrc = WT + (size_t)(nbase + srow) * D_MODEL + sseg;

  v8f acc[8] = {};

  for (int kb = 0; kb < D_MODEL; kb += 32) {
    // ---- stage A (fp32 -> f16) and B into LDS ----
    {
      const v4f x0 = *reinterpret_cast<const v4f*>(xsrc + kb);
      const v4f x1 = *reinterpret_cast<const v4f*>(xsrc + kb + 4);
      const v4f x2 = *reinterpret_cast<const v4f*>(xsrc + kb + 8);
      const v4f x3 = *reinterpret_cast<const v4f*>(xsrc + kb + 12);
      v8h p0, p1;
#pragma unroll
      for (int j = 0; j < 4; ++j) {
        p0[j]     = (_Float16)x0[j];
        p0[4 + j] = (_Float16)x1[j];
        p1[j]     = (_Float16)x2[j];
        p1[4 + j] = (_Float16)x3[j];
      }
      *reinterpret_cast<v8h*>(lA + srow * 32 + sseg)     = p0;
      *reinterpret_cast<v8h*>(lA + srow * 32 + sseg + 8) = p1;

      const v8h b0 = *reinterpret_cast<const v8h*>(wsrc + kb);
      const v8h b1 = *reinterpret_cast<const v8h*>(wsrc + kb + 8);
      *reinterpret_cast<v8h*>(lB + srow * 32 + sseg)     = b0;
      *reinterpret_cast<v8h*>(lB + srow * 32 + sseg + 8) = b1;
    }
    __syncthreads();

    // ---- compute: one A fragment, 8 n-tiles ----
    const v16h a = wmma_ld_a16(lA + (wave * 16 + lmod) * 32, lhalf);
#pragma unroll
    for (int nt = 0; nt < 8; ++nt) {
      const v16h b = *reinterpret_cast<const v16h*>(lB + (nt * 16 + lmod) * 32 + lhalf * 16);
      acc[nt] = __builtin_amdgcn_wmma_f32_16x16x32_f16(false, a, false, b, (short)0,
                                                       acc[nt], false, false);
    }
    __syncthreads();
  }

  // ---- epilogue: bias + phi + store ----
  const int mrow0 = mbase + wave * 16 + lhalf * 8;         // 8 consecutive rows per lane
  const int bb = mrow0 >> 13;
  const int ss = mrow0 & (SEQ - 1);
#pragma unroll
  for (int nt = 0; nt < 8; ++nt) {
    const int nglob = nbase + nt * 16 + lmod;
    const int h = nglob >> 6, d = nglob & 63;
    const float bn = bias[nglob];
    float vals[8];
#pragma unroll
    for (int r = 0; r < 8; ++r) {
      float v = acc[nt][r] + bn;
      if (MODE != 2) v = phi_f(v);
      vals[r] = v;
    }
    if (MODE == 0) {
      _Float16* drow = dst + (((size_t)bb * NUM_HEADS + h) * SEQ) * DEPTH + d;
#pragma unroll
      for (int r = 0; r < 8; ++r)
        drow[(size_t)(ss + r) * DEPTH] = (_Float16)vals[r];
    } else {
      v8h p;
#pragma unroll
      for (int r = 0; r < 8; ++r) p[r] = (_Float16)vals[r];
      *reinterpret_cast<v8h*>(dst + (((size_t)bb * NUM_HEADS + h) * DEPTH + d) * SEQ + ss) = p;
    }
  }
}

// ---------------------------------------------------------------------------
// Kernel 2: partial kv / k_red over one S-chunk.
// Grid (NCHUNK, 32), block 512 = 16 waves; wave -> (dtile, etile).
// k_red via WMMA with B = ones (every accumulator column = row sum).
// The ones-WMMA runs UNCONDITIONALLY in every wave so no EXEC masking can
// surround a WMMA (ISA: WMMA requires EXEC all-ones and is not skipped on
// EXEC==0); only the scalar k_red store is predicated.
// Partials stored as fp32 (contiguous v4f from the C-fragment layout).
// ---------------------------------------------------------------------------
__global__ void lmha_kv(const _Float16* __restrict__ KT, const _Float16* __restrict__ VT,
                        float* __restrict__ kvPart, float* __restrict__ kredPart) {
  const int chunk = blockIdx.x;
  const int bh    = blockIdx.y;
  const int wave  = threadIdx.x >> 5;
  const int lane  = threadIdx.x & 31;
  const int lmod  = lane & 15;
  const int lhalf = lane >> 4;
  const int dt = wave >> 2, et = wave & 3;

  const _Float16* kbase = KT + (size_t)bh * DEPTH * SEQ + (size_t)(dt * 16 + lmod) * SEQ
                             + chunk * SCHUNK;
  const _Float16* vbase = VT + (size_t)bh * DEPTH * SEQ + (size_t)(et * 16 + lmod) * SEQ
                             + chunk * SCHUNK;

  v16h ones;
#pragma unroll
  for (int j = 0; j < 16; ++j) ones[j] = (_Float16)1.0f;

  v8f acc = {}, accs = {};
  for (int s0 = 0; s0 < SCHUNK; s0 += 32) {
    const v16h a = wmma_ld_a16(kbase + s0, lhalf);
    const v16h b = *reinterpret_cast<const v16h*>(vbase + s0 + lhalf * 16);
    acc  = __builtin_amdgcn_wmma_f32_16x16x32_f16(false, a, false, b, (short)0, acc,
                                                  false, false);
    accs = __builtin_amdgcn_wmma_f32_16x16x32_f16(false, a, false, ones, (short)0, accs,
                                                  false, false);
  }

  // kvPart[chunk][bh][e][d]: lane holds fixed e, 8 consecutive d -> two v4f stores
  {
    const int e     = et * 16 + lmod;
    const int dbase = dt * 16 + lhalf * 8;
    float* kvp = kvPart + ((size_t)chunk * 32 + bh) * (DEPTH * DEPTH) + e * DEPTH + dbase;
    v4f lo, hi;
#pragma unroll
    for (int r = 0; r < 4; ++r) { lo[r] = acc[r]; hi[r] = acc[4 + r]; }
    *reinterpret_cast<v4f*>(kvp)     = lo;
    *reinterpret_cast<v4f*>(kvp + 4) = hi;
  }
  if (et == 0 && lmod == 0) {  // column 0 of ones-accumulator = partial k_red
    float* krp = kredPart + ((size_t)chunk * 32 + bh) * DEPTH + dt * 16 + lhalf * 8;
#pragma unroll
    for (int r = 0; r < 8; ++r) krp[r] = accs[r];
  }
}

// ---------------------------------------------------------------------------
// Kernel 2b: reduce the NCHUNK partials -> f16 kvT and kred (+EPS).
// ---------------------------------------------------------------------------
__global__ void lmha_reduce(const float* __restrict__ kvPart,
                            const float* __restrict__ kredPart,
                            _Float16* __restrict__ kvT, _Float16* __restrict__ kred) {
  const int idx = blockIdx.x * blockDim.x + threadIdx.x;
  const int NKV = 32 * DEPTH * DEPTH;   // 131072
  if (idx < NKV) {
    float s = 0.0f;
#pragma unroll
    for (int c = 0; c < NCHUNK; ++c) s += kvPart[(size_t)c * NKV + idx];
    kvT[idx] = (_Float16)s;
  } else if (idx < NKV + 32 * DEPTH) {
    const int j = idx - NKV;
    float s = 0.0f;
#pragma unroll
    for (int c = 0; c < NCHUNK; ++c) s += kredPart[c * (32 * DEPTH) + j];
    kred[j] = (_Float16)(s + EPS_F);
  }
}

// ---------------------------------------------------------------------------
// Kernel 3: out[s][h*64+e] = (qh[s]·kv[:,e]) / (qh[s]·k_red).
// Grid (SEQ/128, 32), block 256 = 8 waves; each wave a 16x64 slab.
// denom via WMMA with B = k_red broadcast -> fragment aligns with outputs.
// ---------------------------------------------------------------------------
__global__ void lmha_out(const _Float16* __restrict__ Qh, const _Float16* __restrict__ kvT,
                         const _Float16* __restrict__ kred, float* __restrict__ out) {
  const int bh    = blockIdx.y;
  const int wave  = threadIdx.x >> 5;
  const int lane  = threadIdx.x & 31;
  const int lmod  = lane & 15;
  const int lhalf = lane >> 4;
  const int stile = blockIdx.x * 8 + wave;

  const _Float16* qrow = Qh  + ((size_t)bh * SEQ + stile * 16 + lmod) * DEPTH;
  const _Float16* kvb  = kvT + (size_t)bh * DEPTH * DEPTH + (size_t)lmod * DEPTH;
  const _Float16* krb  = kred + (size_t)bh * DEPTH;

  v8f acc[4] = {};
  v8f accd = {};
#pragma unroll
  for (int ks = 0; ks < 2; ++ks) {
    const int kb = ks * 32;
    const v16h a  = wmma_ld_a16(qrow + kb, lhalf);
    const v16h kr = *reinterpret_cast<const v16h*>(krb + kb + lhalf * 16);
    accd = __builtin_amdgcn_wmma_f32_16x16x32_f16(false, a, false, kr, (short)0, accd,
                                                  false, false);
#pragma unroll
    for (int et = 0; et < 4; ++et) {
      const v16h b = *reinterpret_cast<const v16h*>(kvb + (size_t)et * 16 * DEPTH + kb +
                                                    lhalf * 16);
      acc[et] = __builtin_amdgcn_wmma_f32_16x16x32_f16(false, a, false, b, (short)0,
                                                       acc[et], false, false);
    }
  }

  const int b = bh >> 3, h = bh & 7;
#pragma unroll
  for (int r = 0; r < 8; ++r) {
    const int s = stile * 16 + lhalf * 8 + r;
    const float inv = 1.0f / accd[r];
    float* orow = out + ((size_t)b * SEQ + s) * D_MODEL + h * DEPTH;
#pragma unroll
    for (int et = 0; et < 4; ++et)
      orow[et * 16 + lmod] = acc[et][r] * inv;
  }
}

// ---------------------------------------------------------------------------
extern "C" void kernel_launch(void* const* d_in, const int* in_sizes, int n_in,
                              void* d_out, int out_size, void* d_ws, size_t ws_size,
                              hipStream_t stream) {
  const float* q  = (const float*)d_in[0];
  const float* k  = (const float*)d_in[1];
  const float* v  = (const float*)d_in[2];
  const float* wq = (const float*)d_in[3];
  const float* bq = (const float*)d_in[4];
  const float* wk = (const float*)d_in[5];
  const float* bk = (const float*)d_in[6];
  const float* wv = (const float*)d_in[7];
  const float* bv = (const float*)d_in[8];
  float* out = (float*)d_out;

  char* ws = (char*)d_ws;
  _Float16* WT   = (_Float16*)ws;                               // 3*512*512 f16
  _Float16* Qh   = (_Float16*)(ws + (size_t)3 * 512 * 512 * 2); // MTOT*512 f16
  _Float16* KT   = Qh + (size_t)MTOT * D_MODEL;
  _Float16* VT   = KT + (size_t)MTOT * D_MODEL;
  _Float16* kvT  = VT + (size_t)MTOT * D_MODEL;                 // 32*64*64 f16
  _Float16* kred = kvT + 32 * DEPTH * DEPTH;                    // 32*64 f16
  float* kvPart   = (float*)(kred + 32 * DEPTH);                // 16*32*64*64 f32
  float* kredPart = kvPart + (size_t)NCHUNK * 32 * DEPTH * DEPTH; // 16*32*64 f32

  lmha_wprep<<<3 * 512 * 512 / 256, 256, 0, stream>>>(wq, wk, wv, WT);

  dim3 gproj(MTOT / 128, D_MODEL / 128, 1);   // (256, 4)
  lmha_proj<0><<<gproj, 256, 0, stream>>>(q, WT,          bq, Qh);
  lmha_proj<1><<<gproj, 256, 0, stream>>>(k, WT + 262144, bk, KT);
  lmha_proj<2><<<gproj, 256, 0, stream>>>(v, WT + 524288, bv, VT);

  dim3 gkv(NCHUNK, 32, 1);
  lmha_kv<<<gkv, 512, 0, stream>>>(KT, VT, kvPart, kredPart);
  lmha_reduce<<<(32 * DEPTH * DEPTH + 32 * DEPTH) / 256, 256, 0, stream>>>(
      kvPart, kredPart, kvT, kred);

  dim3 gout(SEQ / 128, 32, 1);
  lmha_out<<<gout, 256, 0, stream>>>(Qh, kvT, kred, out);
}
